// JODIE_41068477284512
// MI455X (gfx1250) — matrix operand order repaired and checked by reading, not compiled, not run on hardware
//
#include <hip/hip_runtime.h>
#include <hip/hip_bf16.h>

typedef __attribute__((ext_vector_type(16))) _Float16 v16h;
typedef __attribute__((ext_vector_type(8)))  float    v8f;

union Frag16 { v16h h; uint4 u[2]; };

#define E_TOTAL   100000
#define MEM_DIM   256
#define NODE_DIM  128
#define DZ        384
#define HID       384
#define TILE_E    16
#define KSTEPS    (DZ / 32)     // 12
#define NTILES    (HID / 16)    // 24

// ---------------------------------------------------------------------------
// Kernel 0: convert w_lsrc / w_ldst (f32 row-major [HID][DZ]) to f16 in d_ws.
// Row-major [n][k] is exactly the B-fragment-friendly layout (two contiguous 8-half
// K slabs per lane-half).
// ---------------------------------------------------------------------------
__global__ void jodie_convert_weights(const float* __restrict__ wl,
                                      const float* __restrict__ wd,
                                      _Float16* __restrict__ wl16,
                                      _Float16* __restrict__ wd16,
                                      int n) {
    int t = blockIdx.x * blockDim.x + threadIdx.x;
    if (t < n) {
        wl16[t] = (_Float16)wl[t];
        wd16[t] = (_Float16)wd[t];
    }
}

// ---------------------------------------------------------------------------
// Main kernel: one workgroup handles 16 edges (M=16), 256 threads = 8 waves.
// ---------------------------------------------------------------------------
__global__ __launch_bounds__(256)
void jodie_fused(const float* __restrict__ memory_table,
                 const float* __restrict__ last_update,
                 const float* __restrict__ x,
                 const float* __restrict__ batch_t,
                 const float* __restrict__ wp_src,
                 const float* __restrict__ bp_src,
                 const float* __restrict__ wp_dst,
                 const float* __restrict__ bp_dst,
                 const _Float16* __restrict__ wlsrc16,
                 const _Float16* __restrict__ wldst16,
                 const float* __restrict__ b_lsrc,
                 const float* __restrict__ b_ldst,
                 const float* __restrict__ w_final,
                 const float* __restrict__ b_final,
                 const int* __restrict__ n_id,
                 const int* __restrict__ id_mapper,
                 const int* __restrict__ src,
                 const int* __restrict__ pos_dst,
                 const int* __restrict__ neg_dst,
                 float* __restrict__ out_pos,
                 float* __restrict__ out_neg,
                 float* __restrict__ out_msrc,
                 float* __restrict__ out_mpos,
                 int E) {
    __shared__ __align__(16) _Float16 lA[3 * TILE_E * DZ];   // 36 KB f16 A tiles
    __shared__ int   sNode[3][TILE_E];
    __shared__ float sRel [3][TILE_E];
    __shared__ float sPos[TILE_E];
    __shared__ float sNeg[TILE_E];

    const int tid  = threadIdx.x;
    const int lane = tid & 31;
    const int wave = tid >> 5;          // 0..7
    const int lm   = lane & 15;
    const int lh   = lane >> 4;
    const int e0   = blockIdx.x * TILE_E;

    // ---- Phase 1: resolve indices + rel_t ---------------------------------
    if (tid < 48) {
        const int role = tid >> 4;      // 0=src, 1=pos, 2=neg
        const int e    = tid & 15;
        int ge = e0 + e;
        if (ge >= E) ge = E - 1;        // clamp (reads only; writes guarded)
        const int* idxArr = (role == 0) ? src : ((role == 1) ? pos_dst : neg_dst);
        const int node = n_id[id_mapper[idxArr[ge]]];
        sNode[role][e] = node;
        sRel [role][e] = fabsf(last_update[node] - batch_t[ge]);
    }
    if (tid >= 48 && tid < 64) {
        sPos[tid - 48] = 0.0f;
        sNeg[tid - 48] = 0.0f;
    }
    __syncthreads();

    // ---- Phase 2: gather + Jodie scale -> f16 LDS; stream m outputs -------
    // 3 roles * 16 edges * 384 dims = 18432 elements / 256 threads = 72 iters
    #pragma unroll 4
    for (int it = 0; it < 72; ++it) {
        const int idx  = tid + it * 256;
        const int d    = idx % DZ;
        const int re   = idx / DZ;       // 0..47
        const int e    = re & 15;
        const int role = re >> 4;
        const int node = sNode[role][e];
        const int ge   = e0 + e;

        float val;
        if (d < MEM_DIM) {
            val = memory_table[(long long)node * MEM_DIM + d];
            if (ge < E) {
                if (role == 0)      out_msrc[(long long)ge * MEM_DIM + d] = val;
                else if (role == 1) out_mpos[(long long)ge * MEM_DIM + d] = val;
            }
        } else {
            val = x[(long long)node * NODE_DIM + (d - MEM_DIM)];
        }
        const float* wp = (role == 0) ? wp_src : wp_dst;
        const float* bp = (role == 0) ? bp_src : bp_dst;
        const float z = val * (1.0f + sRel[role][e] * wp[d] + bp[d]);
        lA[role * (TILE_E * DZ) + e * DZ + d] = (_Float16)z;
    }
    __syncthreads();

    // ---- Phase 3: WMMA GEMMs ---------------------------------------------
    // Wave w owns N-tiles {w, w+8, w+16}. Accumulate Hs, Hp, Hn.
    v8f accS[3] = {}, accP[3] = {}, accN[3] = {};

    for (int ks = 0; ks < KSTEPS; ++ks) {
        const int k0 = ks * 32;

        // A fragments (16x32 f16): per lane two contiguous 16B LDS slabs.
        Frag16 aS, aP, aN;
        {
            const _Float16* bS = &lA[0 * (TILE_E * DZ) + lm * DZ + k0 + lh * 8];
            const _Float16* bP = &lA[1 * (TILE_E * DZ) + lm * DZ + k0 + lh * 8];
            const _Float16* bN = &lA[2 * (TILE_E * DZ) + lm * DZ + k0 + lh * 8];
            aS.u[0] = *(const uint4*)(bS);      aS.u[1] = *(const uint4*)(bS + 16);
            aP.u[0] = *(const uint4*)(bP);      aP.u[1] = *(const uint4*)(bP + 16);
            aN.u[0] = *(const uint4*)(bN);      aN.u[1] = *(const uint4*)(bN + 16);
        }

        #pragma unroll
        for (int i = 0; i < 3; ++i) {
            const int n = (wave + 8 * i) * 16 + lm;      // B column for this lane
            // B fragments (32x16 f16): two 8-half K slabs per lane-half (k0+8h, k0+16+8h), matching A.
            Frag16 bL, bD;
            const _Float16* pL = wlsrc16 + (long long)n * DZ + k0 + lh * 8;
            const _Float16* pD = wldst16 + (long long)n * DZ + k0 + lh * 8;
            bL.u[0] = *(const uint4*)(pL);  bL.u[1] = *(const uint4*)(pL + 16);
            bD.u[0] = *(const uint4*)(pD);  bD.u[1] = *(const uint4*)(pD + 16);

            accS[i] = __builtin_amdgcn_wmma_f32_16x16x32_f16(
                false, aS.h, false, bL.h, (short)0, accS[i], false, false);
            accP[i] = __builtin_amdgcn_wmma_f32_16x16x32_f16(
                false, aP.h, false, bD.h, (short)0, accP[i], false, false);
            accN[i] = __builtin_amdgcn_wmma_f32_16x16x32_f16(
                false, aN.h, false, bD.h, (short)0, accN[i], false, false);
        }
    }

    // ---- Phase 4: relu + final dot, reduce over n via LDS atomics ---------
    float posSum[8], negSum[8];
    #pragma unroll
    for (int r = 0; r < 8; ++r) { posSum[r] = 0.0f; negSum[r] = 0.0f; }

    #pragma unroll
    for (int i = 0; i < 3; ++i) {
        const int n   = (wave + 8 * i) * 16 + lm;
        const float bias = b_lsrc[n] + b_ldst[n];
        const float wf   = w_final[n];
        #pragma unroll
        for (int r = 0; r < 8; ++r) {
            const float hs = accS[i][r];
            const float hp = hs + accP[i][r] + bias;
            const float hn = hs + accN[i][r] + bias;
            posSum[r] += fmaxf(hp, 0.0f) * wf;
            negSum[r] += fmaxf(hn, 0.0f) * wf;
        }
    }
    #pragma unroll
    for (int r = 0; r < 8; ++r) {
        const int m = r + 8 * lh;       // edge row within tile
        atomicAdd(&sPos[m], posSum[r]);
        atomicAdd(&sNeg[m], negSum[r]);
    }
    __syncthreads();

    if (tid < TILE_E) {
        const int ge = e0 + tid;
        if (ge < E) {
            const float bf = b_final[0];
            out_pos[ge] = sPos[tid] + bf;
            out_neg[ge] = sNeg[tid] + bf;
        }
    }
}

// ---------------------------------------------------------------------------
extern "C" void kernel_launch(void* const* d_in, const int* in_sizes, int n_in,
                              void* d_out, int out_size, void* d_ws, size_t ws_size,
                              hipStream_t stream) {
    const float* memory_table = (const float*)d_in[0];
    const float* last_update  = (const float*)d_in[1];
    const float* x            = (const float*)d_in[2];
    const float* batch_t      = (const float*)d_in[3];
    const float* wp_src       = (const float*)d_in[4];
    const float* bp_src       = (const float*)d_in[5];
    const float* wp_dst       = (const float*)d_in[6];
    const float* bp_dst       = (const float*)d_in[7];
    const float* w_lsrc       = (const float*)d_in[8];
    const float* b_lsrc       = (const float*)d_in[9];
    const float* w_ldst       = (const float*)d_in[10];
    const float* b_ldst       = (const float*)d_in[11];
    const float* w_final      = (const float*)d_in[12];
    const float* b_final      = (const float*)d_in[13];
    const int*   n_id         = (const int*)d_in[14];
    const int*   id_mapper    = (const int*)d_in[15];
    const int*   src          = (const int*)d_in[16];
    const int*   pos_dst      = (const int*)d_in[17];
    const int*   neg_dst      = (const int*)d_in[18];

    const int E = in_sizes[3];           // batch_t length

    // Workspace: f16 copies of the two readout weight matrices (~590 KB).
    const int WN = HID * DZ;             // 147456
    _Float16* wlsrc16 = (_Float16*)d_ws;
    _Float16* wldst16 = wlsrc16 + WN;

    jodie_convert_weights<<<(WN + 255) / 256, 256, 0, stream>>>(
        w_lsrc, w_ldst, wlsrc16, wldst16, WN);

    float* out      = (float*)d_out;
    float* out_pos  = out;                         // [E]
    float* out_neg  = out + E;                     // [E]
    float* out_msrc = out + 2 * (long long)E;      // [E,256]
    float* out_mpos = out_msrc + (long long)E * MEM_DIM;

    const int nblocks = (E + TILE_E - 1) / TILE_E;   // 6250
    jodie_fused<<<nblocks, 256, 0, stream>>>(
        memory_table, last_update, x, batch_t,
        wp_src, bp_src, wp_dst, bp_dst,
        wlsrc16, wldst16, b_lsrc, b_ldst, w_final, b_final,
        n_id, id_mapper, src, pos_dst, neg_dst,
        out_pos, out_neg, out_msrc, out_mpos, E);
}
